// Model_19353122636300
// MI455X (gfx1250) — matrix operand order repaired and checked by reading, not compiled, not run on hardware
//
#include <hip/hip_runtime.h>
#include <hip/hip_bf16.h>
#include <math.h>

// ---------------- problem constants (from reference) ----------------
#define BB     2
#define CCH    96          // C
#define CPCH   192         // CP = 2C
#define HH     96
#define WWD    96
#define HWSZ   (HH*WWD)    // 9216
#define PIXN   (BB*HWSZ)   // 18432
#define NGRP   12          // GROUPS == HEADS
#define KPTS   9           // 3x3 deformable kernel points
#define CG     16          // head dim == group channels
#define GROW   16          // RDB growth
#define COFF   216         // c_off = GROUPS*K*2

typedef __attribute__((ext_vector_type(16))) __bf16 v16bf;
typedef __attribute__((ext_vector_type(8)))  __bf16 v8bf;
typedef __attribute__((ext_vector_type(8)))  float  v8f;

__device__ __forceinline__ float gelu_f(float x) {
    return 0.5f * x * (1.0f + erff(x * 0.70710678118654752440f));
}

// =====================================================================
// 1x1-conv GEMM on the CDNA5 matrix pipe.
//   Out[m,n] = act( sum_k A[m,k]*W[n,k] + bias[n] ) + Res[m,n]
// KPAD compile-time; A fp32 pixel-major, lda == KPAD, K-pad zero-filled
// => unrolled K-loop with no bounds checks, aligned v8f global loads
// (cvt to bf16 in-register). NT output tiles per wave share one A
// fragment (amortizes A traffic + cvt VALU over NT WMMAs); the
// (16*NT) x KPAD bf16 weight tile is staged once in LDS and fragments
// are read back as 16B ds loads. EXEC is all-ones at every WMMA.
// =====================================================================
template<int KPAD, int NT>
__global__ __launch_bounds__(128)
void dattn_wmma_gemm(const float* __restrict__ A,
                     const __bf16* __restrict__ Wp,   // [NPAD, KPAD] bf16, zero-padded
                     const float* __restrict__ bias,
                     const float* __restrict__ Res, int ldr,
                     float* __restrict__ Out, int ldo,
                     int M, int N, int act)
{
    __shared__ __bf16 wtile[16 * NT * KPAD];
    const int tid = threadIdx.x;

    // cooperative copy of the B tile: rows ntile*16*NT..+16*NT of Wp are
    // one contiguous (16*NT)*KPAD block (row-major, zero-padded rows)
    const __bf16* wsrc = Wp + (size_t)blockIdx.y * (16 * NT * KPAD);
    constexpr int CHUNKS = (16 * NT * KPAD) / 8;    // 16-byte chunks
    for (int i = tid; i < CHUNKS; i += 128)
        ((v8bf*)wtile)[i] = ((const v8bf*)wsrc)[i];
    __syncthreads();

    const int wave = tid >> 5;
    const int lane = tid & 31;
    const int hl   = lane >> 4;                 // 16-lane half
    const int l16  = lane & 15;
    const int mtile = blockIdx.x * 4 + wave;
    if (mtile * 16 >= M) return;                // wave-uniform

    const int row = mtile * 16 + l16;           // A row (same for both halves)
    const float* arow = A + (size_t)row * KPAD;
    __builtin_prefetch(arow + 64 * KPAD, 0, 1); // warm next 64-row block

    v8f acc[NT];
    #pragma unroll
    for (int nt = 0; nt < NT; ++nt) acc[nt] = (v8f){};

    #pragma unroll
    for (int kb = 0; kb < KPAD; kb += 32) {
        // A fragment (16-bit A 16x32 layout):
        //   lanes 0-15 : K = kb+[0,8) and kb+16+[0,8)
        //   lanes 16-31: K = kb+8+[0,8) and kb+24+[0,8)
        v8f a0 = *(const v8f*)(arow + kb + 8 * hl);
        v8f a1 = *(const v8f*)(arow + kb + 16 + 8 * hl);
        v16bf afrag;
        #pragma unroll
        for (int t = 0; t < 8; ++t) {
            afrag[t]     = (__bf16)a0[t];
            afrag[8 + t] = (__bf16)a1[t];
        }
        // B fragments (16-bit B 32x16 layout): lane column = l16,
        // lanes 0-15 hold K=kb+[0,8) and kb+16+[0,8), lanes 16-31 hold K=kb+8+[0,8) and kb+24+[0,8)
        #pragma unroll
        for (int nt = 0; nt < NT; ++nt) {
            const __bf16* wl = wtile + (size_t)(nt * 16 + l16) * KPAD + kb + 8 * hl;
            v8bf b0 = *(const v8bf*)(wl);
            v8bf b1 = *(const v8bf*)(wl + 16);
            v16bf bfrag = __builtin_shufflevector(b0, b1,
                            0,1,2,3,4,5,6,7,8,9,10,11,12,13,14,15);
            acc[nt] = __builtin_amdgcn_wmma_f32_16x16x32_bf16(
                          false, afrag, false, bfrag, (short)0, acc[nt], false, false);
        }
    }

    #pragma unroll
    for (int nt = 0; nt < NT; ++nt) {
        const int col = blockIdx.y * (16 * NT) + nt * 16 + l16;
        if (col < N) {
            const float bv = bias ? bias[col] : 0.0f;
            #pragma unroll
            for (int i = 0; i < 8; ++i) {
                // C/D layout: element i -> M = i + 8*half, N = l16
                int m = mtile * 16 + i + 8 * hl;
                float v = acc[nt][i] + bv;
                if (act == 1)      v = (v > 0.0f) ? v : 0.1f * v;   // LeakyReLU(0.1)
                else if (act == 2) v = gelu_f(v);                    // exact GELU
                if (Res) v += Res[(size_t)m * ldr + col];
                Out[(size_t)m * ldo + col] = v;
            }
        }
    }
}

// fp32 OIHW(1x1) [N,K] -> bf16 [NP,KP], zero padded
__global__ void dattn_prep_w(const float* __restrict__ w, __bf16* __restrict__ wp,
                             int N, int K, int KP, int total)
{
    int idx = blockIdx.x * blockDim.x + threadIdx.x;
    if (idx >= total) return;
    int k = idx % KP, n = idx / KP;
    wp[idx] = (__bf16)((n < N && k < K) ? w[(size_t)n * K + k] : 0.0f);
}

// ---------------- layout / elementwise kernels ----------------
__global__ void dattn_pack_pm(const float* __restrict__ src, float* __restrict__ dst,
                              int Cn, int total)
{
    int idx = blockIdx.x * blockDim.x + threadIdx.x;
    if (idx >= total) return;
    int c = idx % Cn, pix = idx / Cn;
    int b = pix / HWSZ, s = pix - b * HWSZ;
    dst[idx] = src[((size_t)b * Cn + c) * HWSZ + s];
}

// concat([f_aligned, r, flows]) -> pixel-major, K padded 194->224 with zeros
__global__ void dattn_build_xcat(const float* __restrict__ fa, const float* __restrict__ r,
                                 const float* __restrict__ fl, float* __restrict__ dst)
{
    int idx = blockIdx.x * blockDim.x + threadIdx.x;
    if (idx >= PIXN * 224) return;
    int c = idx % 224, pix = idx / 224;
    int b = pix / HWSZ, s = pix - b * HWSZ;
    float v = 0.0f;
    if (c < CCH)            v = fa[((size_t)b * CCH + c) * HWSZ + s];
    else if (c < 2 * CCH)   v = r [((size_t)b * CCH + (c - CCH)) * HWSZ + s];
    else if (c < 194)       v = fl[((size_t)b * 2 + (c - 2 * CCH)) * HWSZ + s];
    dst[idx] = v;
}

__global__ void dattn_zero_pad(float* __restrict__ buf, int ld, int c0, int nc)
{
    int idx = blockIdx.x * blockDim.x + threadIdx.x;
    if (idx >= PIXN * nc) return;
    int c = idx % nc, pix = idx / nc;
    buf[(size_t)pix * ld + c0 + c] = 0.0f;
}

// RDB dense layer: 3x3 conv (cin -> 16) + GELU, appended into hbuf slice.
__global__ void dattn_rdb_conv3(const float* __restrict__ in, int ldi, int cin,
                                const float* __restrict__ w, const float* __restrict__ bias,
                                float* __restrict__ out, int ldo, int cofs)
{
    int idx = blockIdx.x * blockDim.x + threadIdx.x;
    if (idx >= PIXN * GROW) return;
    int co  = idx & (GROW - 1);
    int pix = idx >> 4;
    int b = pix / HWSZ, s = pix - b * HWSZ;
    int y = s / WWD, x = s - y * WWD;
    float acc = bias[co];
    for (int ky = 0; ky < 3; ++ky) {
        int yy = y + ky - 1; if (yy < 0 || yy >= HH) continue;
        for (int kx = 0; kx < 3; ++kx) {
            int xx = x + kx - 1; if (xx < 0 || xx >= WWD) continue;
            const float* ip = in + ((size_t)(b * HWSZ + yy * WWD + xx)) * ldi;
            const float* wp = w + ((size_t)co * cin) * 9 + ky * 3 + kx;
            for (int ci = 0; ci < cin; ++ci) acc += ip[ci] * wp[(size_t)ci * 9];
        }
    }
    out[(size_t)pix * ldo + cofs + co] = gelu_f(acc);
}

// off += tile(flows[:, ::-1], 108)
__global__ void dattn_off_add_flow(float* __restrict__ off, const float* __restrict__ fl)
{
    int idx = blockIdx.x * blockDim.x + threadIdx.x;
    if (idx >= PIXN * COFF) return;
    int c = idx % COFF, pix = idx / COFF;
    int b = pix / HWSZ, s = pix - b * HWSZ;
    int fc = 1 - (c & 1);
    off[idx] += fl[((size_t)b * 2 + fc) * HWSZ + s];
}

// Deformable sampling + per-head attention (head == group, hd == cg == 16).
__global__ void dattn_deform_attn(const float* __restrict__ q, const float* __restrict__ kf,
                                  const float* __restrict__ vf, const float* __restrict__ off,
                                  float* __restrict__ out)
{
    int idx = blockIdx.x * blockDim.x + threadIdx.x;
    if (idx >= PIXN * NGRP) return;
    int g = idx % NGRP, pix = idx / NGRP;
    int b = pix / HWSZ, s = pix - b * HWSZ;
    int y = s / WWD, x = s - y * WWD;
    const int gofs = g * CG;

    float qv[CG];
    const float* qp = q + (size_t)pix * CPCH + gofs;
    #pragma unroll
    for (int c = 0; c < CG; ++c) qv[c] = qp[c];

    const float* op = off + (size_t)pix * COFF + g * 2 * KPTS;
    float sy[KPTS], sx[KPTS], lg[KPTS];
    #pragma unroll
    for (int k = 0; k < KPTS; ++k) {
        sy[k] = op[k * 2 + 0] + (float)y + (float)(k / 3 - 1);
        sx[k] = op[k * 2 + 1] + (float)x + (float)(k % 3 - 1);
    }

    float mx = -1e30f;
    for (int k = 0; k < KPTS; ++k) {
        float py = sy[k], px = sx[k];
        float y0f = floorf(py), x0f = floorf(px);
        int y0 = (int)y0f, x0 = (int)x0f;
        float wy = py - y0f, wx = px - x0f;
        float d = 0.0f;
        #pragma unroll
        for (int cor = 0; cor < 4; ++cor) {
            int yi = y0 + (cor >> 1), xi = x0 + (cor & 1);
            float wgt = ((cor >> 1) ? wy : (1.0f - wy)) * ((cor & 1) ? wx : (1.0f - wx));
            if (yi >= 0 && yi < HH && xi >= 0 && xi < WWD) {
                const float* p = kf + ((size_t)(b * HWSZ + yi * WWD + xi)) * CPCH + gofs;
                float dd = 0.0f;
                #pragma unroll
                for (int c = 0; c < CG; ++c) dd += qv[c] * p[c];
                d += wgt * dd;
            }
        }
        lg[k] = 0.25f * d;                  // 1/sqrt(16)
        mx = fmaxf(mx, lg[k]);
    }
    float ssum = 0.0f;
    #pragma unroll
    for (int k = 0; k < KPTS; ++k) { lg[k] = expf(lg[k] - mx); ssum += lg[k]; }
    float inv = 1.0f / ssum;

    float o[CG];
    #pragma unroll
    for (int c = 0; c < CG; ++c) o[c] = 0.0f;
    for (int k = 0; k < KPTS; ++k) {
        float wk = lg[k] * inv;
        float py = sy[k], px = sx[k];
        float y0f = floorf(py), x0f = floorf(px);
        int y0 = (int)y0f, x0 = (int)x0f;
        float wy = py - y0f, wx = px - x0f;
        #pragma unroll
        for (int cor = 0; cor < 4; ++cor) {
            int yi = y0 + (cor >> 1), xi = x0 + (cor & 1);
            float wgt = wk * ((cor >> 1) ? wy : (1.0f - wy)) * ((cor & 1) ? wx : (1.0f - wx));
            if (yi >= 0 && yi < HH && xi >= 0 && xi < WWD) {
                const float* p = vf + ((size_t)(b * HWSZ + yi * WWD + xi)) * CPCH + gofs;
                #pragma unroll
                for (int c = 0; c < CG; ++c) o[c] += wgt * p[c];
            }
        }
    }
    float* dst = out + (size_t)pix * CPCH + gofs;
    #pragma unroll
    for (int c = 0; c < CG; ++c) dst[c] = o[c];
}

// depthwise 3x3, 384 channels, zero pad, no bias
__global__ void dattn_dwconv3(const float* __restrict__ in, const float* __restrict__ w,
                              float* __restrict__ out)
{
    int idx = blockIdx.x * blockDim.x + threadIdx.x;
    if (idx >= PIXN * 4 * CCH) return;
    int ch = idx % (4 * CCH), pix = idx / (4 * CCH);
    int b = pix / HWSZ, s = pix - b * HWSZ;
    int y = s / WWD, x = s - y * WWD;
    float acc = 0.0f;
    for (int ky = 0; ky < 3; ++ky) {
        int yy = y + ky - 1; if (yy < 0 || yy >= HH) continue;
        for (int kx = 0; kx < 3; ++kx) {
            int xx = x + kx - 1; if (xx < 0 || xx >= WWD) continue;
            acc += in[((size_t)(b * HWSZ + yy * WWD + xx)) * (4 * CCH) + ch]
                 * w[(size_t)ch * 9 + ky * 3 + kx];
        }
    }
    out[(size_t)pix * (4 * CCH) + ch] = acc;
}

__global__ void dattn_gate(const float* __restrict__ ydw, float* __restrict__ g)
{
    int idx = blockIdx.x * blockDim.x + threadIdx.x;
    if (idx >= PIXN * CPCH) return;
    int c = idx % CPCH, pix = idx / CPCH;
    float y1 = ydw[(size_t)pix * (4 * CCH) + c];
    float y2 = ydw[(size_t)pix * (4 * CCH) + CPCH + c];
    g[idx] = gelu_f(y1) * y2;
}

// o = proj + ffn -> NCHW
__global__ void dattn_final_out(const float* __restrict__ oproj, const float* __restrict__ ffn,
                                float* __restrict__ dout)
{
    int idx = blockIdx.x * blockDim.x + threadIdx.x;
    if (idx >= PIXN * CCH) return;
    int c = idx % CCH, pix = idx / CCH;
    int b = pix / HWSZ, s = pix - b * HWSZ;
    dout[((size_t)b * CCH + c) * HWSZ + s] = oproj[idx] + ffn[idx];
}

// off_mean = off.reshape(b,108,2,h,w).mean(1)[:, ::-1]  -> appended after o
__global__ void dattn_off_mean(const float* __restrict__ off, float* __restrict__ dout)
{
    int idx = blockIdx.x * blockDim.x + threadIdx.x;
    if (idx >= BB * 2 * HWSZ) return;
    int s = idx % HWSZ, t = idx / HWSZ;
    int oc = t & 1, b = t >> 1;
    const float* op = off + (size_t)(b * HWSZ + s) * COFF + (1 - oc);
    float acc = 0.0f;
    for (int j = 0; j < COFF / 2; ++j) acc += op[j * 2];
    dout[(size_t)BB * CCH * HWSZ + idx] = acc * (1.0f / (float)(COFF / 2));
}

// =====================================================================
extern "C" void kernel_launch(void* const* d_in, const int* in_sizes, int n_in,
                              void* d_out, int out_size, void* d_ws, size_t ws_size,
                              hipStream_t stream)
{
    const float* f    = (const float*)d_in[0];
    const float* fa   = (const float*)d_in[1];
    const float* r    = (const float*)d_in[2];
    const float* fl   = (const float*)d_in[3];
    const float* co0_w = (const float*)d_in[4];  const float* co0_b = (const float*)d_in[5];
    const float* d0_w  = (const float*)d_in[6];  const float* d0_b  = (const float*)d_in[7];
    const float* d1_w  = (const float*)d_in[8];  const float* d1_b  = (const float*)d_in[9];
    const float* d2_w  = (const float*)d_in[10]; const float* d2_b  = (const float*)d_in[11];
    const float* fc_w  = (const float*)d_in[12]; const float* fc_b  = (const float*)d_in[13];
    const float* co3_w = (const float*)d_in[14]; const float* co3_b = (const float*)d_in[15];
    const float* pq_w  = (const float*)d_in[16]; const float* pq_b  = (const float*)d_in[17];
    const float* pk_w  = (const float*)d_in[18]; const float* pk_b  = (const float*)d_in[19];
    const float* pv_w  = (const float*)d_in[20]; const float* pv_b  = (const float*)d_in[21];
    const float* pr_w  = (const float*)d_in[22]; const float* pr_b  = (const float*)d_in[23];
    const float* pin_w  = (const float*)d_in[24];
    const float* dw_w   = (const float*)d_in[25];
    const float* pout_w = (const float*)d_in[26];

    float* ws = (float*)d_ws;
    const size_t P = (size_t)PIXN;
    // ---- long-lived fp32 regions ----
    float* r_pm  = ws;                 // 96P
    float* f_pm  = ws + 96  * P;       // 96P
    float* offb  = ws + 192 * P;       // 216P (alive until off_mean)
    float* oproj = ws + 408 * P;       // 96P  (alive until final residual)
    // ---- bf16 padded weights (tiny) ----
    __bf16* wb   = (__bf16*)(ws + 504 * P);
    __bf16* wb_co0 = wb;               // 96 x 224
    __bf16* wb_fc  = wb + 21504;       // 96 x 160
    __bf16* wb_co3 = wb + 36864;       // 224 x 96 (N padded 216->224)
    __bf16* wb_pq  = wb + 58368;       // 192 x 96
    __bf16* wb_pk  = wb + 76800;
    __bf16* wb_pv  = wb + 95232;
    __bf16* wb_pr  = wb + 113664;      // 96 x 192
    __bf16* wb_pin = wb + 132096;      // 384 x 96
    __bf16* wb_pout= wb + 168960;      // 96 x 192  (end 187392 bf16)
    // ---- phased fp32 region ----
    float* base = ws + 504 * P + 98304;
    // phase 1 (offset network):
    float* xcat = base;                // 224P (K padded 194->224)
    float* hbuf = base + 224 * P;      // 160P (K padded 144->160)
    float* rdbo = base + 384 * P;      // 96P
    // phase 2 (attention):
    float* qbf = base;                 // 192P
    float* kbf = base + 192 * P;       // 192P
    float* vbf = base + 384 * P;       // 192P
    float* oat = base + 576 * P;       // 192P
    // phase 3 (FFN):
    float* ypin = base;                // 384P
    float* ydw  = base + 384 * P;      // 384P
    float* gbuf = base + 768 * P;      // 192P
    float* ffn  = base + 960 * P;      // 96P
    // total ws use: (1560P + 98304) floats ~= 115 MB

    const int TB = 256;
    auto cdiv = [](int a, int b) { return (a + b - 1) / b; };

    // ---- weight prep (fp32 -> zero-padded bf16) ----
    auto prep = [&](const float* w, __bf16* wp, int N, int K, int KP, int NP) {
        int tot = NP * KP;
        dattn_prep_w<<<cdiv(tot, TB), TB, 0, stream>>>(w, wp, N, K, KP, tot);
    };
    prep(co0_w, wb_co0,  96, 194, 224,  96);
    prep(fc_w,  wb_fc,   96, 144, 160,  96);
    prep(co3_w, wb_co3, 216,  96,  96, 224);
    prep(pq_w,  wb_pq,  192,  96,  96, 192);
    prep(pk_w,  wb_pk,  192,  96,  96, 192);
    prep(pv_w,  wb_pv,  192,  96,  96, 192);
    prep(pr_w,  wb_pr,   96, 192, 192,  96);
    prep(pin_w, wb_pin, 384,  96,  96, 384);
    prep(pout_w,wb_pout, 96, 192, 192,  96);

    // ---- layout packs ----
    dattn_pack_pm<<<cdiv(PIXN * CCH, TB), TB, 0, stream>>>(r, r_pm, CCH, PIXN * CCH);
    dattn_pack_pm<<<cdiv(PIXN * CCH, TB), TB, 0, stream>>>(f, f_pm, CCH, PIXN * CCH);
    dattn_build_xcat<<<cdiv(PIXN * 224, TB), TB, 0, stream>>>(fa, r, fl, xcat);
    dattn_zero_pad<<<cdiv(PIXN * 16, TB), TB, 0, stream>>>(hbuf, 160, 144, 16);

    // NT output tiles per wave; grid.y = NPAD/(16*NT)
    #define GEMM(KP, NT, NP, Ain, Wp, Bias, Resp, Ldr, Outp, Ldo, Nn, Act)             \
        dattn_wmma_gemm<KP, NT><<<dim3(PIXN/64, (NP)/(16*(NT))), 128, 0, stream>>>(    \
            Ain, Wp, Bias, Resp, Ldr, Outp, Ldo, PIXN, Nn, Act)

    // ---- offset network ----
    GEMM(224, 2,  96, xcat, wb_co0, co0_b, nullptr, 0, hbuf, 160,  96, 1);
    dattn_rdb_conv3<<<cdiv(PIXN * GROW, TB), TB, 0, stream>>>(hbuf, 160,  96, d0_w, d0_b, hbuf, 160,  96);
    dattn_rdb_conv3<<<cdiv(PIXN * GROW, TB), TB, 0, stream>>>(hbuf, 160, 112, d1_w, d1_b, hbuf, 160, 112);
    dattn_rdb_conv3<<<cdiv(PIXN * GROW, TB), TB, 0, stream>>>(hbuf, 160, 128, d2_w, d2_b, hbuf, 160, 128);
    GEMM(160, 2,  96, hbuf, wb_fc, fc_b, /*Res=x*/hbuf, 160, rdbo, 96, 96, 0);
    GEMM( 96, 2, 224, rdbo, wb_co3, co3_b, nullptr, 0, offb, COFF, COFF, 0);
    dattn_off_add_flow<<<cdiv(PIXN * COFF, TB), TB, 0, stream>>>(offb, fl);

    // ---- q/k/v projections ----
    GEMM( 96, 4, 192, r_pm, wb_pq, pq_b, nullptr, 0, qbf, CPCH, CPCH, 0);
    GEMM( 96, 4, 192, f_pm, wb_pk, pk_b, nullptr, 0, kbf, CPCH, CPCH, 0);
    GEMM( 96, 4, 192, f_pm, wb_pv, pv_b, nullptr, 0, vbf, CPCH, CPCH, 0);

    // ---- deformable sampling + attention ----
    dattn_deform_attn<<<cdiv(PIXN * NGRP, TB), TB, 0, stream>>>(qbf, kbf, vbf, offb, oat);
    GEMM(192, 2,  96, oat, wb_pr, pr_b, nullptr, 0, oproj, CCH, CCH, 0);

    // ---- gated depthwise FFN ----
    GEMM( 96, 4, 384, oproj, wb_pin, nullptr, nullptr, 0, ypin, 4 * CCH, 4 * CCH, 0);
    dattn_dwconv3<<<cdiv(PIXN * 4 * CCH, TB), TB, 0, stream>>>(ypin, dw_w, ydw);
    dattn_gate<<<cdiv(PIXN * CPCH, TB), TB, 0, stream>>>(ydw, gbuf);
    GEMM(192, 2,  96, gbuf, wb_pout, nullptr, nullptr, 0, ffn, CCH, CCH, 0);

    // ---- outputs ----
    dattn_final_out<<<cdiv(PIXN * CCH, TB), TB, 0, stream>>>(oproj, ffn, (float*)d_out);
    dattn_off_mean<<<cdiv(BB * 2 * HWSZ, TB), TB, 0, stream>>>(offb, (float*)d_out);
    #undef GEMM
}
